// ODEForecaster_80985903333558
// MI455X (gfx1250) — matrix operand order repaired and checked by reading, not compiled, not run on hardware
//
#include <hip/hip_runtime.h>

// ---------------------------------------------------------------------------
// Fused ODE(dopri5) -> GRU x2 -> head forecaster for MI455X (gfx1250).
// One persistent workgroup (8 wave32s) per 16-row batch tile; all GEMMs via
// v_wmma_f32_16x16x32_bf16 with f32 accumulation and f32 recurrent state.
// Transcendentals use gfx1250 hardware v_tanh_f32 / v_exp_f32 / v_rcp_f32.
// ---------------------------------------------------------------------------

typedef __attribute__((ext_vector_type(16))) __bf16 v16bf;
typedef __attribute__((ext_vector_type(8)))  float  v8f;

#define NB     1024
#define CT     128      // ODE state dim == GRU input dim
#define TPTS   96
#define HID    128
#define OUTN   3072     // PRED_LEN * CH

union U16b { v16bf v; uint4 q[2]; };

__device__ __forceinline__ v16bf ld2(const __bf16* p0, const __bf16* p1) {
    U16b u;
    u.q[0] = *(const uint4*)p0;
    u.q[1] = *(const uint4*)p1;
    return u.v;
}
// A operand (16x32 bf16): lane m=lane&15, hi=lane>>4; 16 contiguous K at kb*32+hi*16.
__device__ __forceinline__ v16bf loadA16(const __bf16* base, int stride, int kb, int m, int hi) {
    const __bf16* p = base + m * stride + kb * 32 + hi * 16;
    return ld2(p, p + 8);
}
// B operand (32x16 bf16): lane n holds row n of W, 16 contiguous K at kb*32+hi*16.
__device__ __forceinline__ v16bf loadB16(const __bf16* rowbase, int kb, int kh) {
    const __bf16* p = rowbase + kb * 32 + kh * 16;
    return ld2(p, p + 8);
}
__device__ __forceinline__ v16bf cvtB16(const float* p) {
    v16bf v;
#pragma unroll
    for (int i = 0; i < 16; ++i) v[i] = (__bf16)p[i];
    return v;
}
__device__ __forceinline__ v8f wmma_bf16(v16bf a, v16bf b, v8f c) {
    return __builtin_amdgcn_wmma_f32_16x16x32_bf16(false, a, false, b, (short)0, c, false, false);
}

// gfx1250 has V_TANH_F32 as a hardware TRANS op; fall back to libm if the
// builtin is not declared by this toolchain.
__device__ __forceinline__ float fast_tanhf(float x) {
#if __has_builtin(__builtin_amdgcn_tanhf)
    return __builtin_amdgcn_tanhf(x);
#elif __has_builtin(__builtin_amdgcn_tanh_f32)
    return __builtin_amdgcn_tanh_f32(x);
#else
    return tanhf(x);
#endif
}
// Branch-free sigmoid on v_exp_f32 (exp2) + v_rcp_f32: exp2->inf gives rcp->0,
// exp2->0 gives rcp(1)=1, so both saturation limits are correct.
__device__ __forceinline__ float fast_sigm(float x) {
#if __has_builtin(__builtin_amdgcn_exp2f) && __has_builtin(__builtin_amdgcn_rcpf)
    return __builtin_amdgcn_rcpf(1.0f + __builtin_amdgcn_exp2f(-1.44269504f * x));
#else
    return 1.0f / (1.0f + __expf(-x));
#endif
}

// ---------------------------------------------------------------------------
// One PyTorch-semantics GRU layer for this WG's 16-row tile.
// Wave w owns h columns [16w,16w+16); its 3 gate row-slices are j, 128+j, 256+j.
// x part (kb 0-3): B from global bf16 W_ih. h part (kb 4-7): B from LDS W_hh.
// n-gate keeps ih / hh accumulators separate (n = tanh(i_n + r*h_n)).
// ---------------------------------------------------------------------------
__device__ __forceinline__ void gru_layer(
    __bf16* sWhh, __bf16* sXH,
    const __bf16* __restrict__ xsrc,   // [96][16][128] bf16 (this WG)
    const __bf16* __restrict__ wihB,   // [384][128] bf16
    const float*  __restrict__ whhF,   // [384][128] f32 -> LDS
    const float*  __restrict__ bih, const float* __restrict__ bhh,
    __bf16* __restrict__ hout,         // [96][16][128] bf16 or nullptr
    int tid, int ln15, int hi, int j)
{
    for (int i = tid; i < 16 * 136; i += 256) sXH[i] = (__bf16)0.0f;
    for (int i = tid; i < 384 * 128; i += 256)
        sWhh[(i >> 7) * 136 + (i & 127)] = (__bf16)whhF[i];

    const float br  = bih[j] + bhh[j];
    const float bz  = bih[128 + j] + bhh[128 + j];
    const float bni = bih[256 + j];
    const float bnh = bhh[256 + j];
    const __bf16* wR = wihB + (size_t)(j) * CT;
    const __bf16* wZ = wihB + (size_t)(128 + j) * CT;
    const __bf16* wN = wihB + (size_t)(256 + j) * CT;
    float h[8];
#pragma unroll
    for (int r = 0; r < 8; ++r) h[r] = 0.0f;
    __syncthreads();

#pragma unroll 1
    for (int t = 0; t < TPTS; ++t) {
        v8f aR = {}, aZ = {}, aNi = {}, aNh = {};
        const __bf16* xt = xsrc + t * (16 * CT);
        {   // x contribution: hoist A-operand loads so they pipeline
            v16bf A0 = loadA16(xt, CT, 0, ln15, hi);
            v16bf A1 = loadA16(xt, CT, 1, ln15, hi);
            v16bf A2 = loadA16(xt, CT, 2, ln15, hi);
            v16bf A3 = loadA16(xt, CT, 3, ln15, hi);
            aR  = wmma_bf16(A0, loadB16(wR, 0, hi), aR);
            aZ  = wmma_bf16(A0, loadB16(wZ, 0, hi), aZ);
            aNi = wmma_bf16(A0, loadB16(wN, 0, hi), aNi);
            aR  = wmma_bf16(A1, loadB16(wR, 1, hi), aR);
            aZ  = wmma_bf16(A1, loadB16(wZ, 1, hi), aZ);
            aNi = wmma_bf16(A1, loadB16(wN, 1, hi), aNi);
            aR  = wmma_bf16(A2, loadB16(wR, 2, hi), aR);
            aZ  = wmma_bf16(A2, loadB16(wZ, 2, hi), aZ);
            aNi = wmma_bf16(A2, loadB16(wN, 2, hi), aNi);
            aR  = wmma_bf16(A3, loadB16(wR, 3, hi), aR);
            aZ  = wmma_bf16(A3, loadB16(wZ, 3, hi), aZ);
            aNi = wmma_bf16(A3, loadB16(wN, 3, hi), aNi);
        }
        {   // h contribution (LDS weights)
            v16bf A0 = loadA16(sXH, 136, 0, ln15, hi);
            v16bf A1 = loadA16(sXH, 136, 1, ln15, hi);
            v16bf A2 = loadA16(sXH, 136, 2, ln15, hi);
            v16bf A3 = loadA16(sXH, 136, 3, ln15, hi);
            const __bf16* hR = sWhh + (j) * 136;
            const __bf16* hZ = sWhh + (128 + j) * 136;
            const __bf16* hN = sWhh + (256 + j) * 136;
            aR  = wmma_bf16(A0, loadB16(hR, 0, hi), aR);
            aZ  = wmma_bf16(A0, loadB16(hZ, 0, hi), aZ);
            aNh = wmma_bf16(A0, loadB16(hN, 0, hi), aNh);
            aR  = wmma_bf16(A1, loadB16(hR, 1, hi), aR);
            aZ  = wmma_bf16(A1, loadB16(hZ, 1, hi), aZ);
            aNh = wmma_bf16(A1, loadB16(hN, 1, hi), aNh);
            aR  = wmma_bf16(A2, loadB16(hR, 2, hi), aR);
            aZ  = wmma_bf16(A2, loadB16(hZ, 2, hi), aZ);
            aNh = wmma_bf16(A2, loadB16(hN, 2, hi), aNh);
            aR  = wmma_bf16(A3, loadB16(hR, 3, hi), aR);
            aZ  = wmma_bf16(A3, loadB16(hZ, 3, hi), aZ);
            aNh = wmma_bf16(A3, loadB16(hN, 3, hi), aNh);
        }
#pragma unroll
        for (int r = 0; r < 8; ++r) {
            float rg = fast_sigm(aR[r] + br);
            float zg = fast_sigm(aZ[r] + bz);
            float ng = fast_tanhf(aNi[r] + bni + rg * (aNh[r] + bnh));
            h[r] = (1.0f - zg) * ng + zg * h[r];
        }
        __syncthreads();
#pragma unroll
        for (int r = 0; r < 8; ++r) {
            int m = 8 * hi + r;
            __bf16 hb = (__bf16)h[r];
            sXH[m * 136 + j] = hb;
            if (hout) hout[t * (16 * CT) + m * CT + j] = hb;
        }
        __syncthreads();
    }
}

__launch_bounds__(256)
__global__ void ODEForecaster_fused_kernel(
    const float* __restrict__ yl,  const float* __restrict__ yh,
    const float* __restrict__ Wode, const float* __restrict__ bode,
    const __bf16* __restrict__ wih0B, const float* __restrict__ whh0F,
    const float* __restrict__ bih0, const float* __restrict__ bhh0,
    const __bf16* __restrict__ wih1B, const float* __restrict__ whh1F,
    const float* __restrict__ bih1, const float* __restrict__ bhh1,
    const float* __restrict__ W1f, const float* __restrict__ b1f,
    const __bf16* __restrict__ w2B, const float* __restrict__ b2f,
    __bf16* __restrict__ traj, __bf16* __restrict__ h1t,
    float* __restrict__ out)
{
    __shared__ __bf16 sWhh[384 * 136];   // padded recurrent weights (~104.4 KB)
    __shared__ __bf16 sXH [16 * 136];    // y-arg / h tile, padded (4.25 KB)
    __shared__ __bf16 sHID[16 * 72];     // head hidden tile, padded

    const int tid  = threadIdx.x;
    const int wave = tid >> 5;
    const int lane = tid & 31;
    const int ln15 = lane & 15;
    const int hi   = lane >> 4;
    const int j    = wave * 16 + ln15;        // this lane's output column (0..127)
    const int rowbase = blockIdx.x * 16;
    __bf16* traj_wg = traj + (size_t)blockIdx.x * TPTS * 16 * CT;
    __bf16* h1_wg   = h1t  + (size_t)blockIdx.x * TPTS * 16 * CT;

    // ---------------- Phase 1: dopri5 fixed-grid ODE ----------------
    v16bf Bo[4];                              // Wo B-operands, resident all phase
#pragma unroll
    for (int kb = 0; kb < 4; ++kb)
        Bo[kb] = cvtB16(Wode + (size_t)j * CT + kb * 32 + hi * 16);
    const float bo = bode[j];

    float y[8];
#pragma unroll
    for (int r = 0; r < 8; ++r) {
        int m = 8 * hi + r;
        int b = rowbase + m;
        y[r] = (j < 32) ? yl[(size_t)b * 32 + j] : yh[(size_t)b * 96 + (j - 32)];
    }
#pragma unroll
    for (int r = 0; r < 8; ++r) {
        int m = 8 * hi + r;
        __bf16 v = (__bf16)y[r];
        sXH[m * 136 + j] = v;
        traj_wg[m * CT + j] = v;              // t = 0
    }
    __syncthreads();

    const float dt = 1.0f / 95.0f;
    float k1[8], k2[8], k3[8], k4[8], k5[8], k6[8], tmp[8];

    auto feval = [&](float* kd) {
        v16bf A0 = loadA16(sXH, 136, 0, ln15, hi);   // pipeline all A loads
        v16bf A1 = loadA16(sXH, 136, 1, ln15, hi);
        v16bf A2 = loadA16(sXH, 136, 2, ln15, hi);
        v16bf A3 = loadA16(sXH, 136, 3, ln15, hi);
        v8f acc = {};
        acc = wmma_bf16(A0, Bo[0], acc);
        acc = wmma_bf16(A1, Bo[1], acc);
        acc = wmma_bf16(A2, Bo[2], acc);
        acc = wmma_bf16(A3, Bo[3], acc);
#pragma unroll
        for (int r = 0; r < 8; ++r) kd[r] = fast_tanhf(acc[r] + bo);
    };
    auto swrite = [&](const float* a) {
        __syncthreads();                      // all waves done reading sXH
#pragma unroll
        for (int r = 0; r < 8; ++r) sXH[(8 * hi + r) * 136 + j] = (__bf16)a[r];
        __syncthreads();                      // new argument visible
    };

#pragma unroll 1
    for (int t = 1; t < TPTS; ++t) {
        feval(k1);
#pragma unroll
        for (int r = 0; r < 8; ++r) tmp[r] = y[r] + dt * (0.2f * k1[r]);
        swrite(tmp); feval(k2);
#pragma unroll
        for (int r = 0; r < 8; ++r)
            tmp[r] = y[r] + dt * ((3.0f/40.0f) * k1[r] + (9.0f/40.0f) * k2[r]);
        swrite(tmp); feval(k3);
#pragma unroll
        for (int r = 0; r < 8; ++r)
            tmp[r] = y[r] + dt * ((44.0f/45.0f) * k1[r] - (56.0f/15.0f) * k2[r] + (32.0f/9.0f) * k3[r]);
        swrite(tmp); feval(k4);
#pragma unroll
        for (int r = 0; r < 8; ++r)
            tmp[r] = y[r] + dt * ((19372.0f/6561.0f) * k1[r] - (25360.0f/2187.0f) * k2[r]
                                + (64448.0f/6561.0f) * k3[r] - (212.0f/729.0f) * k4[r]);
        swrite(tmp); feval(k5);
#pragma unroll
        for (int r = 0; r < 8; ++r)
            tmp[r] = y[r] + dt * ((9017.0f/3168.0f) * k1[r] - (355.0f/33.0f) * k2[r]
                                + (46732.0f/5247.0f) * k3[r] + (49.0f/176.0f) * k4[r]
                                - (5103.0f/18656.0f) * k5[r]);
        swrite(tmp); feval(k6);
#pragma unroll
        for (int r = 0; r < 8; ++r)
            y[r] = y[r] + dt * ((35.0f/384.0f) * k1[r] + (500.0f/1113.0f) * k3[r]
                              + (125.0f/192.0f) * k4[r] - (2187.0f/6784.0f) * k5[r]
                              + (11.0f/84.0f) * k6[r]);
        swrite(y);
#pragma unroll
        for (int r = 0; r < 8; ++r)
            traj_wg[t * (16 * CT) + (8 * hi + r) * CT + j] = (__bf16)y[r];
    }

    __threadfence(); __syncthreads();

    // ---------------- Phase 2/3: stacked GRU ----------------
    gru_layer(sWhh, sXH, traj_wg, wih0B, whh0F, bih0, bhh0, h1_wg, tid, ln15, hi, j);
    __threadfence(); __syncthreads();
    gru_layer(sWhh, sXH, h1_wg, wih1B, whh1F, bih1, bhh1, nullptr, tid, ln15, hi, j);
    __syncthreads();                          // sXH now holds h2[T-1] (bf16)

    // ---------------- Phase 4: head ----------------
    if (wave < 4) {                           // hidden = gelu(h2 @ W1^T + b1), [16,64]
        const int n = wave * 16 + ln15;
        v16bf A0 = loadA16(sXH, 136, 0, ln15, hi);
        v16bf A1 = loadA16(sXH, 136, 1, ln15, hi);
        v16bf A2 = loadA16(sXH, 136, 2, ln15, hi);
        v16bf A3 = loadA16(sXH, 136, 3, ln15, hi);
        v8f acc = {};
        acc = wmma_bf16(A0, cvtB16(W1f + (size_t)n * CT +  0 + hi * 16), acc);
        acc = wmma_bf16(A1, cvtB16(W1f + (size_t)n * CT + 32 + hi * 16), acc);
        acc = wmma_bf16(A2, cvtB16(W1f + (size_t)n * CT + 64 + hi * 16), acc);
        acc = wmma_bf16(A3, cvtB16(W1f + (size_t)n * CT + 96 + hi * 16), acc);
        float bb = b1f[n];
#pragma unroll
        for (int r = 0; r < 8; ++r) {
            float xg = acc[r] + bb;
            float g = 0.5f * xg * (1.0f + erff(xg * 0.70710678118f));  // exact gelu
            sHID[(8 * hi + r) * 72 + n] = (__bf16)g;
        }
    }
    __syncthreads();

#pragma unroll 1
    for (int i = 0; i < 24; ++i) {            // pred = hidden @ W2^T + b2, [16,3072]
        int nc = (wave * 24 + i) * 16 + ln15;
        v16bf A0 = loadA16(sHID, 72, 0, ln15, hi);
        v16bf A1 = loadA16(sHID, 72, 1, ln15, hi);
        v8f acc = {};
        acc = wmma_bf16(A0, loadB16(w2B + (size_t)nc * 64, 0, hi), acc);
        acc = wmma_bf16(A1, loadB16(w2B + (size_t)nc * 64, 1, hi), acc);
        float bb = b2f[nc];
#pragma unroll
        for (int r = 0; r < 8; ++r)
            out[(size_t)(rowbase + 8 * hi + r) * OUTN + nc] = acc[r] + bb;
    }
}

// f32 -> bf16 weight conversion (W_ih0, W_ih1, W2)
__global__ void cvt_bf16_kernel(const float* __restrict__ s, __bf16* __restrict__ d, int n) {
    int i = blockIdx.x * blockDim.x + threadIdx.x;
    if (i < n) d[i] = (__bf16)s[i];
}

extern "C" void kernel_launch(void* const* d_in, const int* in_sizes, int n_in,
                              void* d_out, int out_size, void* d_ws, size_t ws_size,
                              hipStream_t stream) {
    const float* yl   = (const float*)d_in[1];
    const float* yh   = (const float*)d_in[2];
    const float* Wode = (const float*)d_in[3];
    const float* bode = (const float*)d_in[4];
    const float* Wih0 = (const float*)d_in[5];
    const float* Whh0 = (const float*)d_in[6];
    const float* bih0 = (const float*)d_in[7];
    const float* bhh0 = (const float*)d_in[8];
    const float* Wih1 = (const float*)d_in[9];
    const float* Whh1 = (const float*)d_in[10];
    const float* bih1 = (const float*)d_in[11];
    const float* bhh1 = (const float*)d_in[12];
    const float* W1   = (const float*)d_in[13];
    const float* b1   = (const float*)d_in[14];
    const float* W2   = (const float*)d_in[15];
    const float* b2   = (const float*)d_in[16];
    float* out = (float*)d_out;

    // workspace layout (bytes)
    const size_t WIH0_OFF = 0;                         // 384*128*2 = 98304
    const size_t WIH1_OFF = WIH0_OFF + 98304;          // 98304
    const size_t W2_OFF   = WIH1_OFF + 98304;          // 3072*64*2 = 393216
    const size_t TRAJ_OFF = W2_OFF + 393216;           // 96*1024*128*2 = 25165824
    const size_t H1_OFF   = TRAJ_OFF + 25165824;       // 25165824  (total ~48.6 MB)
    unsigned char* w = (unsigned char*)d_ws;
    __bf16* wih0B = (__bf16*)(w + WIH0_OFF);
    __bf16* wih1B = (__bf16*)(w + WIH1_OFF);
    __bf16* w2B   = (__bf16*)(w + W2_OFF);
    __bf16* traj  = (__bf16*)(w + TRAJ_OFF);
    __bf16* h1t   = (__bf16*)(w + H1_OFF);

    cvt_bf16_kernel<<<(49152  + 255) / 256, 256, 0, stream>>>(Wih0, wih0B, 49152);
    cvt_bf16_kernel<<<(49152  + 255) / 256, 256, 0, stream>>>(Wih1, wih1B, 49152);
    cvt_bf16_kernel<<<(196608 + 255) / 256, 256, 0, stream>>>(W2,   w2B,   196608);

    ODEForecaster_fused_kernel<<<dim3(NB / 16), dim3(256), 0, stream>>>(
        yl, yh, Wode, bode,
        wih0B, Whh0, bih0, bhh0,
        wih1B, Whh1, bih1, bhh1,
        W1, b1, w2B, b2,
        traj, h1t, out);
}
